// BRA_32220844655457
// MI455X (gfx1250) — matrix operand order repaired and checked by reading, not compiled, not run on hardware
//
#include <hip/hip_runtime.h>

// ---------------------------------------------------------------------------
// Types for CDNA5 WMMA (wave32): v_wmma_f32_16x16x32_bf16
// ---------------------------------------------------------------------------
typedef __attribute__((ext_vector_type(16))) __bf16 bf16x16;
typedef __attribute__((ext_vector_type(8)))  __bf16 bf16x8;
typedef __attribute__((ext_vector_type(8)))  float  f32x8;

#define NB 4
#define NN 4000
#define CC 1024
#define DD 1024
#define NREG 5
#define RS 800           // region size (keys per region)

// ---------------------------------------------------------------------------
// CDNA5 async global->LDS copy (16B per lane), tracked by ASYNCcnt.
// LDS byte offset = low 32 bits of the generic pointer (aperture scheme).
// ---------------------------------------------------------------------------
__device__ __forceinline__ void async_copy16(const void* gptr, void* lptr) {
  unsigned lds = (unsigned)(uintptr_t)lptr;
  asm volatile("global_load_async_to_lds_b128 %0, %1, off"
               :: "v"(lds), "v"(gptr) : "memory");
}
__device__ __forceinline__ void wait_async0() {
  asm volatile("s_wait_asynccnt 0x0" ::: "memory");
}
__device__ __forceinline__ void wait_async8() {
  asm volatile("s_wait_asynccnt 0x8" ::: "memory");
}

// ---------------------------------------------------------------------------
// Fragment loaders (per CDNA5 ISA VGPR layouts, wave32)
// A: 16x32 bf16, lane l holds row m=l&15; lanes 0-15: K 0..7 + 16..23,
//    lanes 16-31: K 8..15 + 24..31  -> two contiguous 16B loads.
// B: 32x16 bf16, lane l holds col n=l&15; lanes 0-15: K 0..7 + 16..23,
//    lanes 16-31: K 8..15 + 24..31  -> two contiguous 16B loads.
// Source operand must be k-contiguous with row stride `ld` elements.
// ---------------------------------------------------------------------------
__device__ __forceinline__ bf16x16 load_frag_a(const __bf16* base, int ld) {
  const int lane = threadIdx.x & 31;
  const __bf16* p = base + (size_t)(lane & 15) * ld + ((lane >> 4) << 3);
  bf16x8 lo = *(const bf16x8*)(p);
  bf16x8 hi = *(const bf16x8*)(p + 16);
  bf16x16 r;
#pragma unroll
  for (int i = 0; i < 8; ++i) { r[i] = lo[i]; r[i + 8] = hi[i]; }
  return r;
}

__device__ __forceinline__ bf16x16 load_frag_b(const __bf16* base, int ld) {
  const int lane = threadIdx.x & 31;
  const __bf16* p = base + (size_t)(lane & 15) * ld + ((lane >> 4) << 3);
  bf16x8 lo = *(const bf16x8*)(p);
  bf16x8 hi = *(const bf16x8*)(p + 16);
  bf16x16 r;
#pragma unroll
  for (int i = 0; i < 8; ++i) { r[i] = lo[i]; r[i + 8] = hi[i]; }
  return r;
}

__device__ __forceinline__ f32x8 wmma_bf16(bf16x16 a, bf16x16 b, f32x8 c) {
  return __builtin_amdgcn_wmma_f32_16x16x32_bf16(false, a, false, b, (short)0,
                                                 c, false, false);
}

// ---------------------------------------------------------------------------
// f32 -> bf16 cast
// ---------------------------------------------------------------------------
__global__ void cast_f32_bf16(const float* __restrict__ in,
                              __bf16* __restrict__ out, size_t n) {
  size_t i = (size_t)blockIdx.x * blockDim.x + threadIdx.x;
  size_t stride = (size_t)gridDim.x * blockDim.x;
  for (; i < n; i += stride) out[i] = (__bf16)in[i];
}

// ---------------------------------------------------------------------------
// GEMM: Y[m,n] = sum_k A[m,k] * W[n,k] + bias[n]
// A: M x K bf16 row-major, W: N x K bf16 row-major. Block = 256 threads
// (8 waves), tile 128(M) x 128(N), k-step 64, double-buffered async-to-LDS
// staging of both operands (64 KB LDS). Wave = 64x32 via 4x2 grid of 16x16
// wmma tiles -> 16 wmma per barrier pair.
// ---------------------------------------------------------------------------
template <bool F32OUT>
__global__ __launch_bounds__(256)
void gemm_bf16(const __bf16* __restrict__ A, const __bf16* __restrict__ W,
               const float* __restrict__ bias, void* __restrict__ outp,
               int M, int N, int K) {
  __shared__ __bf16 Abuf[2][128 * 64];  // 2 x 16 KB
  __shared__ __bf16 Bbuf[2][128 * 64];  // 2 x 16 KB

  const int tid  = threadIdx.x;
  const int wave = tid >> 5;
  const int lane = tid & 31;
  const int mtile = blockIdx.x * 128;
  const int ntile = blockIdx.y * 128;

  // Stage copy: 128x64 bf16 = 1024 chunks of 16B per operand; 4 chunks each.
  auto issue_stage = [&](int kk, int buf) {
#pragma unroll
    for (int u = 0; u < 4; ++u) {
      const int c  = tid * 4 + u;
      const int r  = c >> 3;          // 8 chunks per 64-elem row
      const int c8 = (c & 7) * 8;
      async_copy16(A + (size_t)(mtile + r) * K + kk + c8,
                   &Abuf[buf][r * 64 + c8]);
      async_copy16(W + (size_t)(ntile + r) * K + kk + c8,
                   &Bbuf[buf][r * 64 + c8]);
    }
  };

  f32x8 acc[4][2] = {};
  const int S = K / 64;
  issue_stage(0, 0);

  for (int s = 0; s < S; ++s) {
    const int buf = s & 1;
    if (s + 1 < S) {
      issue_stage((s + 1) * 64, buf ^ 1);
      wait_async8();   // stage-s loads done (async loads complete in order)
    } else {
      wait_async0();
    }
    __syncthreads();

    const __bf16* Ab = &Abuf[buf][(wave >> 2) * 64 * 64];
    const __bf16* Bb = &Bbuf[buf][(wave & 3) * 32 * 64];
#pragma unroll
    for (int h = 0; h < 2; ++h) {     // two 32-deep sub-steps
      bf16x16 af[4], bfr[2];
#pragma unroll
      for (int i = 0; i < 4; ++i)
        af[i] = load_frag_a(Ab + i * 16 * 64 + h * 32, 64);
#pragma unroll
      for (int j = 0; j < 2; ++j)
        bfr[j] = load_frag_b(Bb + j * 16 * 64 + h * 32, 64);
#pragma unroll
      for (int i = 0; i < 4; ++i)
#pragma unroll
        for (int j = 0; j < 2; ++j)
          acc[i][j] = wmma_bf16(af[i], bfr[j], acc[i][j]);
    }

    __syncthreads();  // all waves done reading buf before it is re-filled
  }

  // ---- straight-line store epilogue (no runtime branching) ----
  const int mbase = mtile + (wave >> 2) * 64;
  const int nbase = ntile + (wave & 3) * 32;
  const int col = lane & 15;
  const int rhi = (lane >> 4) << 3;
#pragma unroll
  for (int i = 0; i < 4; ++i) {
#pragma unroll
    for (int j = 0; j < 2; ++j) {
      const int n = nbase + j * 16 + col;
      const float bv = bias[n];
#pragma unroll
      for (int v = 0; v < 8; ++v) {
        const size_t m = mbase + i * 16 + v + rhi;
        const float val = acc[i][j][v] + bv;
        if (F32OUT) ((float*)outp)[m * N + n] = val;
        else        ((__bf16*)outp)[m * N + n] = (__bf16)val;
      }
    }
  }
}

// ---------------------------------------------------------------------------
// V transpose per batch: V[b][n][d] (bf16) -> Vt[b][d][n] (bf16)
// Block (32,8), 32x32 LDS tile.
// ---------------------------------------------------------------------------
__global__ void transpose_v(const __bf16* __restrict__ V,
                            __bf16* __restrict__ Vt) {
  __shared__ __bf16 t[32][33];
  const int b = blockIdx.z;
  const int n0 = blockIdx.x * 32;
  const int d0 = blockIdx.y * 32;
  const int tx = threadIdx.x, ty = threadIdx.y;
#pragma unroll
  for (int i = 0; i < 32; i += 8)
    t[ty + i][tx] = V[((size_t)b * NN + n0 + ty + i) * DD + d0 + tx];
  __syncthreads();
#pragma unroll
  for (int i = 0; i < 32; i += 8)
    Vt[(size_t)b * DD * NN + (size_t)(d0 + ty + i) * NN + n0 + tx] =
        t[tx][ty + i];
}

// ---------------------------------------------------------------------------
// Region attention: one block per (16-query tile, batch). 8 waves.
// Q tile (16x1024 bf16, 32 KB) staged once into LDS via async copies.
// Per region g: S = Q_tile @ K_g^T (WMMA -> LDS f32), per-row softmax over
// the 800 region keys, P (bf16, LDS), then acc += P @ V_g (A-frags from LDS,
// B-frags from Vt global, prefetched one step ahead). ctx written as bf16.
// ---------------------------------------------------------------------------
__global__ __launch_bounds__(256)
void attn_kernel(const __bf16* __restrict__ Qb, const __bf16* __restrict__ Kb,
                 const __bf16* __restrict__ Vt, __bf16* __restrict__ ctx) {
  __shared__ __bf16 Qs[16 * DD];  // 32768 B
  __shared__ float  S[16 * RS];   // 51200 B
  __shared__ __bf16 P[16 * RS];   // 25600 B

  const int b   = blockIdx.y;
  const int q0  = blockIdx.x * 16;
  const int tid = threadIdx.x;
  const int wave = tid >> 5;
  const int lane = tid & 31;

  // ---- stage Q tile into LDS: 16*1024*2B = 2048 x 16B chunks, 8/thread ----
  {
    const __bf16* Qrow = Qb + ((size_t)b * NN + q0) * DD;
#pragma unroll
    for (int u = 0; u < 8; ++u) {
      const int c  = tid + u * 256;   // chunk 0..2047
      const int r  = c >> 7;          // 128 chunks per 1024-elem row
      const int c8 = (c & 127) * 8;
      async_copy16(Qrow + (size_t)r * DD + c8, &Qs[r * DD + c8]);
    }
    wait_async0();
    __syncthreads();
  }

  f32x8 acc[8] = {};  // wave owns cols [wave*128, wave*128+128)

  for (int g = 0; g < NREG; ++g) {
    // ---- phase A: scores S[16][800] = Q_tile @ K_g^T ----
    for (int kt = wave; kt < RS / 16; kt += 8) {
      const __bf16* Krow = Kb + ((size_t)b * NN + g * RS + kt * 16) * DD;
      if (kt + 8 < RS / 16) {
        // pull next key tile of this wave toward the caches
        __builtin_prefetch(Krow + (size_t)8 * 16 * DD + lane * 64, 0, 0);
      }
      f32x8 s = {};
      for (int kk = 0; kk < DD; kk += 32) {
        bf16x16 a  = load_frag_a(&Qs[kk], DD);
        bf16x16 bb = load_frag_b(Krow + kk, DD);
        s = wmma_bf16(a, bb, s);
      }
      const int col = kt * 16 + (lane & 15);
      const int rhi = (lane >> 4) << 3;
#pragma unroll
      for (int v = 0; v < 8; ++v) S[(v + rhi) * RS + col] = s[v];
    }
    __syncthreads();

    // ---- phase B: per-row softmax over 800 keys ----
    {
      const int r  = tid >> 4;          // row 0..15
      const int c0 = (tid & 15) * 50;   // 50 cols per thread
      float mx = -3.0e38f;
      for (int j = 0; j < 50; ++j) mx = fmaxf(mx, S[r * RS + c0 + j]);
#pragma unroll
      for (int o = 8; o >= 1; o >>= 1) mx = fmaxf(mx, __shfl_xor(mx, o, 32));
      float sum = 0.f;
      for (int j = 0; j < 50; ++j) {
        float e = __expf(S[r * RS + c0 + j] - mx);
        S[r * RS + c0 + j] = e;
        sum += e;
      }
#pragma unroll
      for (int o = 8; o >= 1; o >>= 1) sum += __shfl_xor(sum, o, 32);
      const float rinv = 1.0f / sum;
      for (int j = 0; j < 50; ++j)
        P[r * RS + c0 + j] = (__bf16)(S[r * RS + c0 + j] * rinv);
    }
    __syncthreads();

    // ---- phase C: acc += P @ V_g  (Vt[b][d][n], k-contiguous per d-row) ----
    const __bf16* Vg = Vt + (size_t)b * DD * NN + g * RS;
    for (int kk = 0; kk < RS; kk += 32) {
      if (kk + 32 < RS) {
        // prefetch next k-panel of this wave's Vt columns
        __builtin_prefetch(Vg + (size_t)(wave * 128 + (lane & 15) * 8) * NN +
                               kk + 32, 0, 0);
      }
      bf16x16 a = load_frag_a(&P[0] + kk, RS);
#pragma unroll
      for (int nt = 0; nt < 8; ++nt) {
        const int cn = wave * 128 + nt * 16;
        bf16x16 bb = load_frag_b(Vg + (size_t)cn * NN + kk, NN);
        acc[nt] = wmma_bf16(a, bb, acc[nt]);
      }
    }
    __syncthreads();
  }

  // ---- store context (bf16) ----
  const int col = lane & 15;
  const int rhi = (lane >> 4) << 3;
#pragma unroll
  for (int nt = 0; nt < 8; ++nt) {
    const int n = wave * 128 + nt * 16 + col;
#pragma unroll
    for (int v = 0; v < 8; ++v)
      ctx[((size_t)b * NN + q0 + v + rhi) * DD + n] = (__bf16)acc[nt][v];
  }
}

// ---------------------------------------------------------------------------
// Host-side launcher
// ---------------------------------------------------------------------------
extern "C" void kernel_launch(void* const* d_in, const int* in_sizes, int n_in,
                              void* d_out, int out_size, void* d_ws,
                              size_t ws_size, hipStream_t stream) {
  const float* x  = (const float*)d_in[0];
  const float* Wq = (const float*)d_in[1];
  const float* bq = (const float*)d_in[2];
  const float* Wk = (const float*)d_in[3];
  const float* bk = (const float*)d_in[4];
  const float* Wv = (const float*)d_in[5];
  const float* bv = (const float*)d_in[6];
  const float* Wo = (const float*)d_in[7];
  const float* bo = (const float*)d_in[8];
  float* out = (float*)d_out;

  const int M = NB * NN;  // 16000
  char* ws = (char*)d_ws;
  size_t off = 0;
  auto alloc = [&](size_t bytes) -> void* {
    void* p = ws + off;
    off += (bytes + 255) & ~(size_t)255;
    return p;
  };

  __bf16* xb  = (__bf16*)alloc((size_t)M * CC * 2);
  __bf16* Wqb = (__bf16*)alloc((size_t)DD * CC * 2);
  __bf16* Wkb = (__bf16*)alloc((size_t)DD * CC * 2);
  __bf16* Wvb = (__bf16*)alloc((size_t)DD * CC * 2);
  __bf16* Wob = (__bf16*)alloc((size_t)DD * DD * 2);
  __bf16* Qb  = (__bf16*)alloc((size_t)M * DD * 2);
  __bf16* Kb  = (__bf16*)alloc((size_t)M * DD * 2);
  __bf16* Vb  = (__bf16*)alloc((size_t)M * DD * 2);
  __bf16* Vt  = (__bf16*)alloc((size_t)M * DD * 2);
  __bf16* ctx = xb;  // x_bf16 is dead after the QKV GEMMs; reuse it

  // ---- casts ----
  cast_f32_bf16<<<4096, 256, 0, stream>>>(x, xb, (size_t)M * CC);
  cast_f32_bf16<<<1024, 256, 0, stream>>>(Wq, Wqb, (size_t)DD * CC);
  cast_f32_bf16<<<1024, 256, 0, stream>>>(Wk, Wkb, (size_t)DD * CC);
  cast_f32_bf16<<<1024, 256, 0, stream>>>(Wv, Wvb, (size_t)DD * CC);
  cast_f32_bf16<<<1024, 256, 0, stream>>>(Wo, Wob, (size_t)DD * DD);

  // ---- QKV projections (M=16000, N=1024, K=1024) ----
  dim3 ggrid(M / 128, DD / 128);
  gemm_bf16<false><<<ggrid, 256, 0, stream>>>(xb, Wqb, bq, Qb, M, DD, CC);
  gemm_bf16<false><<<ggrid, 256, 0, stream>>>(xb, Wkb, bk, Kb, M, DD, CC);
  gemm_bf16<false><<<ggrid, 256, 0, stream>>>(xb, Wvb, bv, Vb, M, DD, CC);

  // ---- V transpose per batch ----
  transpose_v<<<dim3(NN / 32, DD / 32, NB), dim3(32, 8), 0, stream>>>(Vb, Vt);

  // ---- region attention ----
  attn_kernel<<<dim3(NN / 16, NB), 256, 0, stream>>>(Qb, Kb, Vt, ctx);

  // ---- output projection -> f32 ----
  gemm_bf16<true><<<ggrid, 256, 0, stream>>>(ctx, Wob, bo, out, M, DD, DD);
}
